// KANLayer_36258113913304
// MI455X (gfx1250) — matrix operand order repaired and checked by reading, not compiled, not run on hardware
//
#include <hip/hip_runtime.h>

typedef __attribute__((ext_vector_type(16))) __bf16 v16bf;
typedef __attribute__((ext_vector_type(8)))  float  v8f;
typedef __attribute__((ext_vector_type(4)))  int    v4i;

#define N_SAMP 4096
#define N_INF  512
#define N_OUTF 512
#define KTOT   6144   // 512 inputs * 12 features (1 silu + 11 basis)
#define NKT    192    // KTOT / 32
#define LDA    40     // LDS row stride in halves (32 + 8 pad -> conflict-free b128)

#ifndef USE_ASYNC_LDS
#if defined(__has_builtin)
#if __has_builtin(__builtin_amdgcn_global_load_async_to_lds_b128)
#define USE_ASYNC_LDS 1
#endif
#endif
#endif
#ifndef USE_ASYNC_LDS
#define USE_ASYNC_LDS 0
#endif

__device__ __forceinline__ unsigned short f2bf(float f) {
  unsigned u = __float_as_uint(f);
  u += 0x7FFFu + ((u >> 16) & 1u);   // round-to-nearest-even
  return (unsigned short)(u >> 16);
}

#if USE_ASYNC_LDS
typedef __attribute__((address_space(1))) v4i* gas_v4i_p;
typedef __attribute__((address_space(3))) v4i* las_v4i_p;

__device__ __forceinline__ void async_b128(const void* g, void* l) {
  // generic LDS pointer carries the LDS byte offset in its low 32 bits
  __builtin_amdgcn_global_load_async_to_lds_b128(
      (gas_v4i_p)(unsigned long long)g,
      (las_v4i_p)(unsigned int)(unsigned long long)l,
      0, 0);
}

__device__ __forceinline__ void wait_async_le3() {
#if __has_builtin(__builtin_amdgcn_s_wait_asynccnt)
  __builtin_amdgcn_s_wait_asynccnt(3);
#else
  asm volatile("s_wait_asynccnt 3" ::: "memory");
#endif
}
#endif

// ---------------------------------------------------------------------------
// Kernel 1: fuse weights into transposed bf16 matrix Wt[o][kk], kk = i*12 + j
//   j == 0      -> w_silu[i][o]
//   j == 1..11  -> w_sp[i][o] * C[j-1][i][o]
// ---------------------------------------------------------------------------
__global__ __launch_bounds__(256) void kan_pack_w(
    const float* __restrict__ C, const float* __restrict__ w_silu,
    const float* __restrict__ w_sp, unsigned short* __restrict__ Wt)
{
  int idx = blockIdx.x * 256 + threadIdx.x;      // o*KTOT + kk
  int o  = idx / KTOT;
  int kk = idx - o * KTOT;
  int i  = kk / 12;
  int r  = kk - i * 12;
  float v;
  if (r == 0) v = w_silu[i * N_OUTF + o];
  else        v = w_sp[i * N_OUTF + o] * C[((r - 1) * N_INF + i) * N_OUTF + o];
  Wt[idx] = f2bf(v);
}

// ---------------------------------------------------------------------------
// Kernel 2: features F[n][i*12 + j] (bf16, row-major 4096 x 6144)
//   Cox-de Boor degree-3 on the analytic uniform extended knot vector,
//   t_j = (j-3)*0.25 - 1, j = 0..14  (matches reference _make_grid()).
// ---------------------------------------------------------------------------
__global__ __launch_bounds__(256) void kan_features(
    const float* __restrict__ x, unsigned short* __restrict__ F)
{
  int idx = blockIdx.x * 256 + threadIdx.x;      // n*N_INF + i
  float xv = x[idx];
  const float h = 0.25f;
  float b[14];
#pragma unroll
  for (int j = 0; j < 14; ++j) {
    float t0 = (float)(j - 3) * h - 1.0f;
    float t1 = (float)(j - 2) * h - 1.0f;
    b[j] = (xv >= t0 && xv < t1) ? 1.0f : 0.0f;
  }
#pragma unroll
  for (int d = 1; d <= 3; ++d) {
    float inv = 1.0f / ((float)d * h);
#pragma unroll
    for (int j = 0; j < 14 - d; ++j) {
      float tj = (float)(j - 3) * h - 1.0f;          // t_j
      float tr = (float)(j - 2 + d) * h - 1.0f;      // t_{j+d+1}
      b[j] = (xv - tj) * inv * b[j] + (tr - xv) * inv * b[j + 1];
    }
  }
  float si = xv / (1.0f + __expf(-xv));              // silu

  unsigned short hs[12];
  hs[0] = f2bf(si);
#pragma unroll
  for (int j = 0; j < 11; ++j) hs[1 + j] = f2bf(b[j]);

  unsigned p[6];
#pragma unroll
  for (int k = 0; k < 6; ++k)
    p[k] = (unsigned)hs[2 * k] | ((unsigned)hs[2 * k + 1] << 16);

  uint2* d2 = (uint2*)(F + (size_t)idx * 12);        // 24B per thread, 8B aligned
  d2[0] = make_uint2(p[0], p[1]);
  d2[1] = make_uint2(p[2], p[3]);
  d2[2] = make_uint2(p[4], p[5]);
}

// ---------------------------------------------------------------------------
// Kernel 3: out(4096x512) = F(4096x6144) * Wt^T(6144x512)  via WMMA bf16.
// Block tile 128(M) x 64(N), 4 waves (128 thr), each wave 32x64 (2x4 tiles).
// Triple-buffered LDS, async DMA staging at prefetch distance 2 so each
// stage has a full iteration of WMMA work to hide under (async loads
// complete in order => s_wait_asynccnt 3 releases buffer kt+1).
// ---------------------------------------------------------------------------
__global__ __launch_bounds__(128) void kan_gemm(
    const unsigned short* __restrict__ F, const unsigned short* __restrict__ Wt,
    float* __restrict__ out)
{
  __shared__ unsigned short As[3][128 * LDA];
  __shared__ unsigned short Bs[3][64 * LDA];

  const int nb = blockIdx.x;               // 0..7   (64 output cols each)
  const int mb = blockIdx.y;               // 0..31  (128 rows each)
  const int t  = threadIdx.x;
  const int lane = t & 31, l15 = lane & 15, hi = lane >> 4;
  const int wr = t >> 5;                   // wave 0..3 -> rows wr*32..wr*32+31

  const unsigned short* Ag = F  + (size_t)(mb * 128) * KTOT;
  const unsigned short* Bg = Wt + (size_t)(nb * 64)  * KTOT;

  const int srow = t >> 2;                 // 0..31
  const int scol = (t & 3) * 8;            // half-offset inside 32-half tile row

  v8f acc[2][4] = {};

#if USE_ASYNC_LDS
  auto stage = [&](int kt, int buf) {
    const size_t ko = (size_t)kt * 32 + scol;
#pragma unroll
    for (int q = 0; q < 4; ++q)            // A: 128 rows, 32 per pass
      async_b128(Ag + (size_t)(srow + q * 32) * KTOT + ko,
                 &As[buf][(srow + q * 32) * LDA + scol]);
#pragma unroll
    for (int q = 0; q < 2; ++q)            // B: 64 rows, 32 per pass
      async_b128(Bg + (size_t)(srow + q * 32) * KTOT + ko,
                 &Bs[buf][(srow + q * 32) * LDA + scol]);
  };

  stage(0, 0);
  stage(1, 1);
  wait_async_le3();                        // oldest stage (buf0) complete
  __syncthreads();
#else
  uint4 pa[4], pb[2];
  auto gload = [&](int kt) {
    const size_t ko = (size_t)kt * 32 + scol;
#pragma unroll
    for (int q = 0; q < 4; ++q)
      pa[q] = *(const uint4*)(Ag + (size_t)(srow + q * 32) * KTOT + ko);
#pragma unroll
    for (int q = 0; q < 2; ++q)
      pb[q] = *(const uint4*)(Bg + (size_t)(srow + q * 32) * KTOT + ko);
  };
  auto sstore = [&](int buf) {
#pragma unroll
    for (int q = 0; q < 4; ++q)
      *(uint4*)(&As[buf][(srow + q * 32) * LDA + scol]) = pa[q];
#pragma unroll
    for (int q = 0; q < 2; ++q)
      *(uint4*)(&Bs[buf][(srow + q * 32) * LDA + scol]) = pb[q];
  };

  gload(0);
  sstore(0);
  __syncthreads();
#endif

  int cur = 0;
  for (int kt = 0; kt < NKT; ++kt) {
    const bool more = (kt + 1) < NKT;

#if USE_ASYNC_LDS
    if (kt + 2 < NKT) {
      int nxt2 = cur + 2; if (nxt2 >= 3) nxt2 -= 3;
      stage(kt + 2, nxt2);                 // DMA overlaps this iteration's WMMA
    }
#else
    if (more) gload(kt + 1);
#endif

    // A fragment (16x32, ISA 7.12.2): lane m holds K hi*8..+7 and 16+hi*8..+7
    union { v16bf v; uint4 u[2]; } af[2], bfv[4];
#pragma unroll
    for (int mt = 0; mt < 2; ++mt) {
      const unsigned short* p = &As[cur][(wr * 32 + mt * 16 + l15) * LDA + hi * 8];
      af[mt].u[0] = *(const uint4*)p;
      af[mt].u[1] = *(const uint4*)(p + 16);
    }
    // B fragment (32x16): lane n holds K hi*8..+7 and 16+hi*8..+7
#pragma unroll
    for (int nt = 0; nt < 4; ++nt) {
      const unsigned short* p = &Bs[cur][(nt * 16 + l15) * LDA + hi * 8];
      bfv[nt].u[0] = *(const uint4*)p;
      bfv[nt].u[1] = *(const uint4*)(p + 16);
    }

#pragma unroll
    for (int mt = 0; mt < 2; ++mt)
#pragma unroll
      for (int nt = 0; nt < 4; ++nt)
        acc[mt][nt] = __builtin_amdgcn_wmma_f32_16x16x32_bf16(
            false, af[mt].v, false, bfv[nt].v, (short)0, acc[mt][nt],
            false, false);

#if USE_ASYNC_LDS
    if (more) {
      wait_async_le3();                    // stage(kt+1) done (in-order)
      __syncthreads();                     // visible to all waves
    }
#else
    __syncthreads();
    if (more) {
      sstore(cur ^ 1);
      __syncthreads();
    }
#endif

#if USE_ASYNC_LDS
    ++cur; if (cur == 3) cur = 0;
#else
    cur ^= 1;
#endif
  }

  // C/D layout: lanes 0-15 -> M = r, lanes 16-31 -> M = 8 + r, N = lane&15
#pragma unroll
  for (int mt = 0; mt < 2; ++mt) {
    const int rowbase = mb * 128 + wr * 32 + mt * 16 + hi * 8;
#pragma unroll
    for (int nt = 0; nt < 4; ++nt) {
      const int col = nb * 64 + nt * 16 + l15;
#pragma unroll
      for (int r = 0; r < 8; ++r)
        out[(size_t)(rowbase + r) * N_OUTF + col] = acc[mt][nt][r];
    }
  }
}

// ---------------------------------------------------------------------------
extern "C" void kernel_launch(void* const* d_in, const int* in_sizes, int n_in,
                              void* d_out, int out_size, void* d_ws, size_t ws_size,
                              hipStream_t stream) {
  (void)in_sizes; (void)n_in; (void)out_size; (void)ws_size;
  const float* x      = (const float*)d_in[0];
  // d_in[1] = grid: redundant (uniform analytic knots reproduced in-kernel)
  const float* C      = (const float*)d_in[2];
  const float* w_silu = (const float*)d_in[3];
  const float* w_sp   = (const float*)d_in[4];
  float* out = (float*)d_out;

  unsigned short* Wt = (unsigned short*)d_ws;                                   // 6 MiB
  unsigned short* F  = (unsigned short*)((char*)d_ws + (size_t)KTOT * N_OUTF * 2);

  kan_pack_w  <<<(N_OUTF * KTOT) / 256, 256, 0, stream>>>(C, w_silu, w_sp, Wt);
  kan_features<<<(N_SAMP * N_INF) / 256, 256, 0, stream>>>(x, F);
  kan_gemm    <<<dim3(N_OUTF / 64, N_SAMP / 128), 128, 0, stream>>>(F, Wt, out);
}
